// RPN_31138512896523
// MI455X (gfx1250) — matrix operand order repaired and checked by reading, not compiled, not run on hardware
//
#include <hip/hip_runtime.h>

typedef __attribute__((ext_vector_type(16))) __bf16 v16bf;
typedef __attribute__((ext_vector_type(8)))  __bf16 v8bf;
typedef __attribute__((ext_vector_type(8)))  float  v8f;

// GEMM dims: M = 8*128*128 = 131072 positions, N = 256 out channels, K = 2304
// K ordering: k = j*256 + c, j = ky*3 + kx  (so a 32-wide K chunk has constant j)
#define KTOT   2304
#define KCH    32
#define NKC    72          // 2304 / 32
#define TILE_M 64
#define NSCORE (8*4*128*128)   // 524288
#define NPOS   (8*128*128)     // 131072

#if defined(__HIP_DEVICE_COMPILE__) && __has_builtin(__builtin_amdgcn_global_load_async_to_lds_b128)
#define HAVE_ASYNC_LDS 1
typedef int v4i __attribute__((vector_size(16)));
typedef __attribute__((address_space(1))) v4i glob_v4i;   // global int4
typedef __attribute__((address_space(3))) v4i lds_v4i;    // LDS int4
#endif

__device__ __forceinline__ void wait_async0() {
#if defined(HAVE_ASYNC_LDS)
#if __has_builtin(__builtin_amdgcn_s_wait_asynccnt)
  __builtin_amdgcn_s_wait_asynccnt(0);
#else
  asm volatile("s_wait_asynccnt 0x0" ::: "memory");
#endif
#endif
}

// ---------------------------------------------------------------------------
// Prep: w1 (O=256, C=256, 3, 3) f32  ->  bf16, fragment-ordered per K-chunk.
// Flat layout: (((kc*16 + tn)*32 + lane)*16 + e)
//   col o      = tn*16 + (lane&15)
//   klocal     = (lane>>4)*16 + e            (contiguous k order, same as the A fragments)
//   k          = kc*32 + klocal ; c = k&255 ; j = k>>8 ; ky=j/3 ; kx=j%3
// ---------------------------------------------------------------------------
__global__ __launch_bounds__(256) void prep_w_kernel(const float* __restrict__ w1,
                                                     __bf16* __restrict__ wbf) {
  int g = blockIdx.x * 256 + threadIdx.x;
  if (g >= KTOT * 256) return;
  int e  = g & 15;
  int ln = (g >> 4) & 31;
  int tn = (g >> 9) & 15;
  int kc = g >> 13;
  int o  = tn * 16 + (ln & 15);
  int klocal = (ln >> 4) * 16 + e;
  int k  = kc * KCH + klocal;
  int c  = k & 255;
  int j  = k >> 8;
  int ky = j / 3, kx = j - 3 * ky;
  wbf[g] = (__bf16)w1[o * 2304 + c * 9 + ky * 3 + kx];
}

// ---------------------------------------------------------------------------
// Fused conv3x3 (implicit GEMM, bf16 WMMA) + bias + 1x1 cls/reg epilogue.
// Block: 64 consecutive positions (same b,y; x in [x0,x0+63]), all 256 N.
// 8 waves = 2(M) x 4(N); each wave 32x64 = 2x4 WMMA tiles.
// Software pipelined: gather A(k+1) to regs + async B(k+1) to LDS during
// the WMMAs of chunk k.
// ---------------------------------------------------------------------------
__global__ __launch_bounds__(256) void conv_rpn_kernel(
    const float* __restrict__ x,   const __bf16* __restrict__ wbf,
    const float* __restrict__ b1,  const float* __restrict__ wc,
    const float* __restrict__ bc,  const float* __restrict__ wr,
    const float* __restrict__ br,  float* __restrict__ scores,
    float* __restrict__ reg_all)
{
  // Phase-overlaid LDS arena (CU-mode 64KB budget):
  //   loop phase:     As [0,4K)   Bs0 [4K,20K)   Bs1 [20K,36K)
  //   epilogue phase: fs [0,32K)  wlds [32K,52K)  blds [+80B]
  __shared__ __align__(32) char smem[32768 + 20480 + 128];
  __bf16* As  = (__bf16*)smem;
  __bf16* Bs0 = (__bf16*)(smem + 4096);
  __bf16* Bs1 = (__bf16*)(smem + 4096 + 16384);
  __bf16* fs  = (__bf16*)smem;
  float*  wlds = (float*)(smem + 32768);
  float*  blds = (float*)(smem + 32768 + 20480);

  const int t    = threadIdx.x;
  const int lane = t & 31;
  const int wave = t >> 5;
  const int wy   = wave >> 2;    // 0..1  (M)
  const int wx   = wave & 3;     // 0..3  (N)

  const int m0 = blockIdx.x * TILE_M;
  const int b  = m0 >> 14;
  const int y  = (m0 >> 7) & 127;
  const int x0 = m0 & 127;

  // Per-thread A-writer constants: thread t owns As flat indices t*8 .. t*8+7.
  // A fragment layout (contiguous k order, as in B):
  //   m_local = tm*16 + (lane&15); kb=(lane>>4)*8;
  //   klocal  = (lane>>4)*16 + e
  int klv[8], xbv[8];
  #pragma unroll
  for (int i = 0; i < 8; ++i) {
    int f  = t * 8 + i;
    int e  = f & 15;
    int ln = (f >> 4) & 31;
    int tm = f >> 9;
    int ml = tm * 16 + (ln & 15);
    int kb = (ln >> 4) * 8;
    klv[i] = (ln >> 4) * 16 + e;
    xbv[i] = x0 + ml - 1;                  // + kx gives input x coordinate
  }

  // Branchless gather of one A chunk into a packed v8bf register.
  auto gather_a = [&](int kc) -> v8bf {
    const int j  = kc >> 3;                // filter tap, constant per chunk
    const int ky = j / 3;
    const int kx = j - 3 * ky;
    const int yi = y + ky - 1;
    const int yok = ((unsigned)yi < 128u) ? 1 : 0;
    const int c0 = (kc & 7) << 5;          // base input channel of chunk
    const int rowbase = (b << 22) + yi * 128;
    v8bf p;
    #pragma unroll
    for (int i = 0; i < 8; ++i) {
      int cc = c0 + klv[i];
      int xi = xbv[i] + kx;
      int ok = yok & (((unsigned)xi < 128u) ? 1 : 0);
      int addr = rowbase + (cc << 14) + xi;
      addr = ok ? addr : 0;                // safe clamp, no exec divergence
      float v = x[addr];
      p[i] = (__bf16)(ok ? v : 0.0f);
    }
    return p;
  };

#if defined(HAVE_ASYNC_LDS)
  auto stage_b_async = [&](int kc, __bf16* Bsb) {
    const __bf16* src = wbf + (size_t)kc * (256 * KCH);
    #pragma unroll
    for (int i = 0; i < 4; ++i) {
      int r = t + 256 * i;                 // 1024 rows of 16B
      __builtin_amdgcn_global_load_async_to_lds_b128(
          (glob_v4i*)(src + r * 8), (lds_v4i*)(Bsb + r * 8), 0, 0);
    }
  };
#else
  auto stage_b_sync = [&](int kc, __bf16* Bsb) {
    const unsigned* src = (const unsigned*)(wbf + (size_t)kc * (256 * KCH));
    unsigned* dst = (unsigned*)Bsb;
    #pragma unroll
    for (int i = 0; i < 16; ++i) dst[t + 256 * i] = src[t + 256 * i];
  };
#endif

  v8f acc[2][4] = {};

  // ---- pipeline prologue ----
  v8bf apack = gather_a(0);
#if defined(HAVE_ASYNC_LDS)
  stage_b_async(0, Bs0);
#endif

  for (int kc = 0; kc < NKC; ++kc) {
    __bf16* Bsb = (kc & 1) ? Bs1 : Bs0;
    __bf16* Bsn = (kc & 1) ? Bs0 : Bs1;

    __syncthreads();                       // previous compute done; As free
    *(v8bf*)&As[t * 8] = apack;            // one b128 store per thread
#if defined(HAVE_ASYNC_LDS)
    wait_async0();                         // B chunk kc landed in Bsb
#else
    stage_b_sync(kc, Bsb);
#endif
    __syncthreads();                       // LDS tiles visible to all waves

    if (kc + 1 < NKC) {
#if defined(HAVE_ASYNC_LDS)
      stage_b_async(kc + 1, Bsn);          // overlaps with WMMAs below
#endif
      apack = gather_a(kc + 1);            // global latency hidden by WMMAs
    }

    // ---- WMMA: 2 A-frags x 4 B-frags per wave ----
    v16bf a0 = *(const v16bf*)&As[((wy * 2 + 0) * 32 + lane) * 16];
    v16bf a1 = *(const v16bf*)&As[((wy * 2 + 1) * 32 + lane) * 16];
    #pragma unroll
    for (int n = 0; n < 4; ++n) {
      v16bf bf = *(const v16bf*)&Bsb[((wx * 4 + n) * 32 + lane) * 16];
      acc[0][n] = __builtin_amdgcn_wmma_f32_16x16x32_bf16(
          false, a0, false, bf, (short)0, acc[0][n], false, false);
      acc[1][n] = __builtin_amdgcn_wmma_f32_16x16x32_bf16(
          false, a1, false, bf, (short)0, acc[1][n], false, false);
    }
  }

  __syncthreads();                         // all compute done; arena reusable

  // ---- stage 1x1 weights into LDS (rows: 4 cls(ch 2a+1), 16 reg) ----
  for (int idx = t; idx < 20 * 256; idx += 256) {
    int o = idx >> 8, ch = idx & 255;
    wlds[idx] = (o < 4) ? wc[(2 * o + 1) * 256 + ch] : wr[(o - 4) * 256 + ch];
  }
  if (t < 20) blds[t] = (t < 4) ? bc[2 * t + 1] : br[t - 4];

  // ---- store f tile (+ bias) to LDS as bf16 ----
  // C/D layout: vgpr i, lanes<16: M=i,   N=lane ; lanes>=16: M=i+8, N=lane-16
  const int hi  = lane >> 4;
  const int nn0 = lane & 15;
  #pragma unroll
  for (int tm = 0; tm < 2; ++tm) {
    #pragma unroll
    for (int tn = 0; tn < 4; ++tn) {
      int N = (wx * 4 + tn) * 16 + nn0;
      float bias = b1[N];
      #pragma unroll
      for (int i = 0; i < 8; ++i) {
        int M = (wy * 2 + tm) * 16 + hi * 8 + i;
        fs[M * 256 + N] = (__bf16)(acc[tm][tn][i] + bias);
      }
    }
  }
  __syncthreads();

  // ---- fused 1x1 convs: 4 cls scores + 16 reg per position ----
  for (int task = t; task < TILE_M * 20; task += 256) {
    int m = task / 20;
    int o = task - m * 20;
    float s = blds[o];
    for (int ch = 0; ch < 256; ++ch) s += (float)fs[m * 256 + ch] * wlds[o * 256 + ch];
    if (o < 4) {
      scores[(((b * 4 + o) * 128 + y) << 7) + x0 + m] = s;
    } else {
      reg_all[(m0 + m) * 16 + (o - 4)] = s;
    }
  }
}

// ---------------------------------------------------------------------------
// Ordered selection of first 1024 indices with score >= 0 (row-major order).
// ---------------------------------------------------------------------------
__global__ __launch_bounds__(256) void count_kernel(const float* __restrict__ scores,
                                                    int* __restrict__ counts) {
  __shared__ int sh[256];
  int t = threadIdx.x;
  int base = blockIdx.x * 1024 + t * 4;
  int c = 0;
  #pragma unroll
  for (int i = 0; i < 4; ++i) c += (scores[base + i] >= 0.0f) ? 1 : 0;
  sh[t] = c;
  __syncthreads();
  for (int s = 128; s > 0; s >>= 1) { if (t < s) sh[t] += sh[t + s]; __syncthreads(); }
  if (t == 0) counts[blockIdx.x] = sh[0];
}

__global__ __launch_bounds__(512) void scan_kernel(const int* __restrict__ counts,
                                                   int* __restrict__ offsets,
                                                   int* __restrict__ total) {
  __shared__ int sh[512];
  int t = threadIdx.x;
  int v = counts[t];
  sh[t] = v;
  __syncthreads();
  for (int d = 1; d < 512; d <<= 1) {
    int add = (t >= d) ? sh[t - d] : 0;
    __syncthreads();
    sh[t] += add;
    __syncthreads();
  }
  offsets[t] = sh[t] - v;     // exclusive prefix
  if (t == 511) total[0] = sh[511];
}

__global__ __launch_bounds__(256) void select_kernel(const float* __restrict__ scores,
                                                     const int* __restrict__ offsets,
                                                     int* __restrict__ sel) {
  __shared__ int sh[256];
  int t = threadIdx.x;
  int base = blockIdx.x * 1024 + t * 4;
  int msk[4]; int c = 0;
  #pragma unroll
  for (int i = 0; i < 4; ++i) { msk[i] = scores[base + i] >= 0.0f; c += msk[i]; }
  sh[t] = c;
  __syncthreads();
  for (int d = 1; d < 256; d <<= 1) {
    int add = (t >= d) ? sh[t - d] : 0;
    __syncthreads();
    sh[t] += add;
    __syncthreads();
  }
  int rank = offsets[blockIdx.x] + sh[t] - c;
  #pragma unroll
  for (int i = 0; i < 4; ++i) {
    if (msk[i]) { if (rank < 1024) sel[rank] = base + i; rank++; }
  }
}

__global__ __launch_bounds__(256) void final_kernel(
    const float* __restrict__ scores, const float* __restrict__ reg_all,
    const int* __restrict__ sel, const int* __restrict__ total,
    const int* __restrict__ hptr, const int* __restrict__ wptr,
    float* __restrict__ out)
{
  int slot = blockIdx.x * 256 + threadIdx.x;
  int nv = total[0];
  if (nv > 1024) nv = 1024;
  float cls = 0.f, r0 = 0.f, r1 = 0.f, r2 = 0.f, r3 = 0.f;
  float c0 = 0.f, c1 = 0.f, c2 = 0.f, c3 = 0.f;
  if (slot < nv) {
    int idx = sel[slot];
    int xi = idx & 127;
    int tt = idx >> 7;
    int yi = tt & 127;  tt >>= 7;
    int a  = tt & 3;
    int bb = tt >> 2;
    int pos = (bb << 14) + (yi << 7) + xi;
    cls = scores[idx];
    r0 = reg_all[pos * 16 + a * 4 + 0];
    r1 = reg_all[pos * 16 + a * 4 + 1];
    r2 = reg_all[pos * 16 + a * 4 + 2];
    r3 = reg_all[pos * 16 + a * 4 + 3];
    float wf = (float)wptr[0], hf = (float)hptr[0];
    float offx = fmodf(wf - 35.0f, 16.0f) * 0.5f + 17.5f;
    float offy = fmodf(hf - 35.0f, 16.0f) * 0.5f + 17.5f;
    c0 = (float)xi * 16.0f + offx;
    c1 = (float)yi * 16.0f + offy;
    const float AW[4] = {64.f, 64.f, 128.f, 128.f};
    const float AH[4] = {64.f, 128.f, 128.f, 256.f};
    c2 = AW[a];
    c3 = AH[a];
  }
  out[slot] = cls;
  float* ro = out + 1024;
  ro[slot * 4 + 0] = r0; ro[slot * 4 + 1] = r1;
  ro[slot * 4 + 2] = r2; ro[slot * 4 + 3] = r3;
  float* bo = out + 1024 + 4096;
  bo[slot * 4 + 0] = c0; bo[slot * 4 + 1] = c1;
  bo[slot * 4 + 2] = c2; bo[slot * 4 + 3] = c3;
}

// ---------------------------------------------------------------------------
extern "C" void kernel_launch(void* const* d_in, const int* in_sizes, int n_in,
                              void* d_out, int out_size, void* d_ws, size_t ws_size,
                              hipStream_t stream) {
  const float* x  = (const float*)d_in[0];
  const float* w1 = (const float*)d_in[1];
  const float* b1 = (const float*)d_in[2];
  const float* wc = (const float*)d_in[3];
  const float* bc = (const float*)d_in[4];
  const float* wr = (const float*)d_in[5];
  const float* br = (const float*)d_in[6];
  const int*   hp = (const int*)d_in[7];
  const int*   wp = (const int*)d_in[8];
  float* out = (float*)d_out;

  // workspace layout (bytes)
  char* ws = (char*)d_ws;
  __bf16* wbf     = (__bf16*)(ws + 0);                       // 2304*256*2 = 1,179,648
  float*  scores  = (float*)(ws + 1179648);                  // 524288*4   = 2,097,152
  float*  reg_all = (float*)(ws + 1179648 + 2097152);        // 131072*16*4= 8,388,608
  char*   ctrl    = ws + 1179648 + 2097152 + 8388608;
  int* counts  = (int*)(ctrl);          // 512 ints
  int* offsets = (int*)(ctrl + 2048);   // 512 ints
  int* total   = (int*)(ctrl + 4096);   // 1 int
  int* sel     = (int*)(ctrl + 4352);   // 1024 ints

  prep_w_kernel<<<(KTOT * 256 + 255) / 256, 256, 0, stream>>>(w1, wbf);
  conv_rpn_kernel<<<NPOS / TILE_M, 256, 0, stream>>>(
      x, wbf, b1, wc, bc, wr, br, scores, reg_all);
  count_kernel<<<NSCORE / 1024, 256, 0, stream>>>(scores, counts);
  scan_kernel<<<1, 512, 0, stream>>>(counts, offsets, total);
  select_kernel<<<NSCORE / 1024, 256, 0, stream>>>(scores, offsets, sel);
  final_kernel<<<4, 256, 0, stream>>>(scores, reg_all, sel, total, hp, wp, out);
}
